// MultiHeadAttention_18425409700485
// MI455X (gfx1250) — matrix operand order repaired and checked by reading, not compiled, not run on hardware
//
#include <hip/hip_runtime.h>
#include <hip/hip_bf16.h>

typedef _Float16 half_t;
typedef __attribute__((ext_vector_type(16))) _Float16 v16h;
typedef __attribute__((ext_vector_type(8)))  _Float16 v8h;
typedef __attribute__((ext_vector_type(4)))  _Float16 v4h;
typedef __attribute__((ext_vector_type(8)))  float    v8f;
typedef __attribute__((ext_vector_type(4)))  unsigned int u32x4;
typedef __attribute__((ext_vector_type(4)))  int i32x4;
typedef __attribute__((ext_vector_type(8)))  int i32x8;

// Problem constants
#define S_LEN 2048
#define D_EMB 1024
#define N_HEAD 16
#define HEAD_DIM 64
#define MROWS 4096              // B*S
#define LDSP 40                 // padded LDS row stride (halfs): 16B-aligned, bank-conflict-free

__device__ __forceinline__ v8f wmma16(v16h a, v16h b, v8f c) {
    return __builtin_amdgcn_wmma_f32_16x16x32_f16(false, a, false, b, (short)0, c, false, false);
}

__device__ __forceinline__ v16h cat8(v8h lo, v8h hi) {
    return __builtin_shufflevector(lo, hi, 0,1,2,3,4,5,6,7,8,9,10,11,12,13,14,15);
}

__device__ __forceinline__ float red_max16(float v) {
    v = fmaxf(v, __shfl_xor(v, 1, 32));
    v = fmaxf(v, __shfl_xor(v, 2, 32));
    v = fmaxf(v, __shfl_xor(v, 4, 32));
    v = fmaxf(v, __shfl_xor(v, 8, 32));
    return v;
}
__device__ __forceinline__ float red_sum16(float v) {
    v += __shfl_xor(v, 1, 32);
    v += __shfl_xor(v, 2, 32);
    v += __shfl_xor(v, 4, 32);
    v += __shfl_xor(v, 8, 32);
    return v;
}

// ---------------------------------------------------------- TDM 2D tile load
// D# per CDNA5 ISA: group0 = {count, lds_addr, global_addr, type=2},
// group1 = {data_size=2B, pad(interval=16DW, amount=4DW) -> LDS row stride 40 halfs,
//           tensor dims/strides, tile dims}. 2D tile => groups 2/3 zero.
__device__ __forceinline__ void tdm_load_tile_2d(unsigned lds_off, const half_t* gptr,
                                                 unsigned tensor_w, unsigned tensor_h,
                                                 unsigned tile_w, unsigned tile_h,
                                                 unsigned stride_elems) {
    unsigned long long ga = (unsigned long long)(size_t)gptr;
    u32x4 g0;
    g0[0] = 1u;                                                  // count=1, no gather
    g0[1] = lds_off;                                             // LDS byte address
    g0[2] = (unsigned)(ga & 0xFFFFFFFFu);                        // global_addr lo
    g0[3] = (unsigned)((ga >> 32) & 0x1FFFFFFu) | (2u << 30);    // global_addr hi | type=2
    i32x8 g1;
    g1[0] = (int)((1u << 16) | (1u << 20) | (3u << 22) | (3u << 25)); // 2B, pad en, 16DW/4DW
    g1[1] = (int)((tensor_w & 0xFFFFu) << 16);                   // tensor_dim0[15:0]
    g1[2] = (int)((tensor_w >> 16) | ((tensor_h & 0xFFFFu) << 16));
    g1[3] = (int)((tensor_h >> 16) | (tile_w << 16));            // tile_dim0
    g1[4] = (int)tile_h;                                         // tile_dim1 (tile_dim2=0)
    g1[5] = (int)stride_elems;                                   // tensor_dim0_stride lo
    g1[6] = 0;
    g1[7] = 0;
    i32x4 z4 = {0, 0, 0, 0};
#if defined(__clang_major__) && (__clang_major__ >= 23)
    i32x8 z8 = {};
    __builtin_amdgcn_tensor_load_to_lds(g0, g1, z4, z4, z8, 0);
#else
    __builtin_amdgcn_tensor_load_to_lds(g0, g1, z4, z4, 0);
#endif
}

// ---------------------------------------------------------------- f32 -> f16
__global__ __launch_bounds__(256) void cvt4_kernel(const float4* __restrict__ in,
                                                   v4h* __restrict__ out, int n4) {
    int i = blockIdx.x * 256 + threadIdx.x;
    if (i < n4) {
        float4 f = in[i];
        v4h h;
        h[0] = (half_t)f.x; h[1] = (half_t)f.y; h[2] = (half_t)f.z; h[3] = (half_t)f.w;
        out[i] = h;
    }
}

// ------------------------------- weight transpose + convert: W[k][n] -> Wt[n][k] f16
__global__ __launch_bounds__(256) void transpose_cvt_kernel(const float* __restrict__ W,
                                                            half_t* __restrict__ Wt) {
    __shared__ float tile[32][33];
    const int tx = threadIdx.x & 31, ty = threadIdx.x >> 5;   // 32 x 8
    const int kb = blockIdx.y * 32, nb = blockIdx.x * 32;
    #pragma unroll
    for (int i = 0; i < 32; i += 8)
        tile[ty + i][tx] = W[(size_t)(kb + ty + i) * D_EMB + nb + tx];
    __syncthreads();
    #pragma unroll
    for (int i = 0; i < 32; i += 8)
        Wt[(size_t)(nb + ty + i) * D_EMB + kb + tx] = (half_t)tile[tx][ty + i];
}

// ------------------------------------------------------- tiled WMMA f16 GEMM
// C[M,N] = A[M,K] @ Wt[N,K]^T + bias[N]
// Block: 128 threads (4 waves). Block tile 128(M) x 64(N); wave tile 32 x 64.
// B tile (64n x 32k, LDS stride 40) streamed by the Tensor Data Mover, double-buffered.
// A fragments straight from global (no inter-wave reuse).
// mode 0: f16 row-major (Q,K) | mode 1: f16 V^T layout vt[b][h][dh][s] | mode 2: f32 + bias
__global__ __launch_bounds__(128)
void gemm_wmma_kernel(const half_t* __restrict__ A, const half_t* __restrict__ Wt,
                      const float* __restrict__ bias, void* __restrict__ Cout,
                      int M, int N, int K, int mode) {
    __shared__ half_t Bs[2][64 * LDSP];

    const int tid  = threadIdx.x;
    const int lane = tid & 31;
    const int wave = tid >> 5;
    const int rown = lane & 15;
    const int hf   = lane >> 4;
    const int m0 = blockIdx.y * 128;
    const int n0 = blockIdx.x * 64;

    v8f acc[2][4] = {};      // 2 row-tiles x 4 col-groups
    const int nsteps = K / 32;
    const half_t* btile = Wt + (size_t)n0 * K;

    if (wave == 0)
        tdm_load_tile_2d((unsigned)(size_t)&Bs[0][0], btile, K, N, 32, 64, K);

    const half_t* arow0 = A + (size_t)(m0 + wave * 32 + rown) * K;
    const half_t* arow1 = arow0 + (size_t)16 * K;

    for (int kk = 0; kk < nsteps; ++kk) {
        const int k0 = kk * 32;
        if (wave == 0) {
            if (kk + 1 < nsteps) {
                tdm_load_tile_2d((unsigned)(size_t)&Bs[(kk + 1) & 1][0],
                                 btile + (kk + 1) * 32, K, N, 32, 64, K);
                __builtin_amdgcn_s_wait_tensorcnt(1);   // current tile done, next in flight
            } else {
                __builtin_amdgcn_s_wait_tensorcnt(0);
            }
        }
        __syncthreads();                                 // tile kk visible to all waves

        // A fragments from global (ISA 7.12.2 16-bit A map)
        v16h a0 = cat8(*(const v8h*)(arow0 + k0 + hf * 8),
                       *(const v8h*)(arow0 + k0 + 16 + hf * 8));
        v16h a1 = cat8(*(const v8h*)(arow1 + k0 + hf * 8),
                       *(const v8h*)(arow1 + k0 + 16 + hf * 8));

        const half_t* Bsc = &Bs[kk & 1][0];
        #pragma unroll
        for (int g = 0; g < 4; ++g) {
            int bn = g * 16 + rown;
            v16h bf = cat8(*(const v8h*)&Bsc[bn * LDSP + hf * 8],
                           *(const v8h*)&Bsc[bn * LDSP + 16 + hf * 8]);
            acc[0][g] = wmma16(a0, bf, acc[0][g]);
            acc[1][g] = wmma16(a1, bf, acc[1][g]);
        }
        __syncthreads();                                 // buffer consumed before overwrite
    }

    #pragma unroll
    for (int rt = 0; rt < 2; ++rt) {
        #pragma unroll
        for (int g = 0; g < 4; ++g) {
            int col = n0 + g * 16 + rown;
            float bvv = bias[col];
            #pragma unroll
            for (int r = 0; r < 8; ++r) {
                int gm = m0 + wave * 32 + rt * 16 + r + 8 * hf;
                float val = acc[rt][g][r] + bvv;
                if (mode == 0) {
                    ((half_t*)Cout)[(size_t)gm * N + col] = (half_t)val;
                } else if (mode == 1) {
                    int bb = gm >> 11, s = gm & (S_LEN - 1);
                    int hh = col >> 6, dh = col & (HEAD_DIM - 1);
                    ((half_t*)Cout)[(((size_t)(bb * N_HEAD + hh) * HEAD_DIM + dh) << 11) + s] = (half_t)val;
                } else {
                    ((float*)Cout)[(size_t)gm * N + col] = val;
                }
            }
        }
    }
}

// ----------------------------------------------------- flash attention (f16)
// One wave per (b, h, 16-row q-block); 32 keys/iter; online softmax; writes the
// reference's buggy merged layout M[b][h*128 + dh*2 + (s>>10)][s & 1023].
__global__ __launch_bounds__(128)
void flash_attn_kernel(const half_t* __restrict__ qf, const half_t* __restrict__ kf,
                       const half_t* __restrict__ vt, half_t* __restrict__ mOut) {
    __shared__ half_t P[4][16 * LDSP];

    const int tid  = threadIdx.x;
    const int lane = tid & 31;
    const int wave = tid >> 5;
    const int rown = lane & 15;
    const int hf   = lane >> 4;

    const int wid = blockIdx.x * 4 + wave;
    const int qb  = wid & 127;
    const int h   = (wid >> 7) & (N_HEAD - 1);
    const int b   = wid >> 11;
    const int q0  = qb * 16;

    const half_t* qrow = qf + (size_t)(b * S_LEN + q0 + rown) * D_EMB + h * HEAD_DIM;
    v16h aq0 = cat8(*(const v8h*)(qrow +      hf * 16), *(const v8h*)(qrow + 8 + hf * 16));
    v16h aq1 = cat8(*(const v8h*)(qrow + 32 + hf * 16), *(const v8h*)(qrow + 40 + hf * 16));

    v8f acc[4] = {};
    float mrow[8], lrow[8];
    #pragma unroll
    for (int r = 0; r < 8; ++r) { mrow[r] = -3.0e38f; lrow[r] = 0.0f; }

    half_t* Pw = &P[wave][0];
    const half_t* vbase = vt + ((size_t)(b * N_HEAD + h) * HEAD_DIM << 11);

    const int nb = (q0 + 47) >> 5;
    for (int t = 0; t < nb; ++t) {
        const int t0 = t * 32;
        const half_t* krow0 = kf + (size_t)(b * S_LEN + t0 + rown) * D_EMB + h * HEAD_DIM;
        const half_t* krow1 = krow0 + 16 * D_EMB;
        v16h bk0lo = *(const v16h*)(krow0 + hf * 16);
        v16h bk0hi = *(const v16h*)(krow0 + 32 + hf * 16);
        v16h bk1lo = *(const v16h*)(krow1 + hf * 16);
        v16h bk1hi = *(const v16h*)(krow1 + 32 + hf * 16);

        v8f s0 = {}, s1 = {};
        s0 = wmma16(aq0, bk0lo, s0);  s0 = wmma16(aq1, bk0hi, s0);
        s1 = wmma16(aq0, bk1lo, s1);  s1 = wmma16(aq1, bk1hi, s1);

        float p0[8], p1[8];
        #pragma unroll
        for (int r = 0; r < 8; ++r) {
            int qg = q0 + r + 8 * hf;
            float v0 = (s0[r] + ((t0 + rown)      > qg ? -1.0e9f : 0.0f)) * 0.03125f;
            float v1 = (s1[r] + ((t0 + 16 + rown) > qg ? -1.0e9f : 0.0f)) * 0.03125f;
            float mx = red_max16(fmaxf(v0, v1));
            float mn = fmaxf(mrow[r], mx);
            p0[r] = __expf(v0 - mn);
            p1[r] = __expf(v1 - mn);
            float sum = red_sum16(p0[r] + p1[r]);
            float sc = __expf(mrow[r] - mn);
            lrow[r] = lrow[r] * sc + sum;
            mrow[r] = mn;
            #pragma unroll
            for (int g = 0; g < 4; ++g) acc[g][r] *= sc;
        }

        #pragma unroll
        for (int r = 0; r < 8; ++r) {
            Pw[(r + 8 * hf) * LDSP + rown]      = (half_t)p0[r];
            Pw[(r + 8 * hf) * LDSP + 16 + rown] = (half_t)p1[r];
        }
        asm volatile("s_wait_dscnt 0x0" ::: "memory");   // DS in-order per wave (ISA 7.3)
        v16h ap = cat8(*(const v8h*)&Pw[rown * LDSP + hf * 16],
                       *(const v8h*)&Pw[rown * LDSP + hf * 16 + 8]);

        #pragma unroll
        for (int g = 0; g < 4; ++g) {
            const half_t* vrow = vbase + ((size_t)(g * 16 + rown) << 11) + t0 + hf * 16;
            v16h bv = *(const v16h*)vrow;
            acc[g] = wmma16(ap, bv, acc[g]);
        }
    }

    float inv[8];
    #pragma unroll
    for (int r = 0; r < 8; ++r) inv[r] = 1.0f / lrow[r];
    #pragma unroll
    for (int g = 0; g < 4; ++g) {
        int dh = g * 16 + rown;
        #pragma unroll
        for (int r = 0; r < 8; ++r) {
            int s = q0 + r + 8 * hf;
            size_t row = (size_t)b * S_LEN + h * 128 + dh * 2 + (s >> 10);
            mOut[(row << 10) + (s & 1023)] = (half_t)(acc[g][r] * inv[r]);
        }
    }
}

// ---------------------------------------------------------------------------
extern "C" void kernel_launch(void* const* d_in, const int* in_sizes, int n_in,
                              void* d_out, int out_size, void* d_ws, size_t ws_size,
                              hipStream_t stream) {
    (void)in_sizes; (void)n_in; (void)out_size; (void)ws_size;
    const float* query = (const float*)d_in[0];
    const float* key   = (const float*)d_in[1];
    const float* value = (const float*)d_in[2];
    const float* Wq = (const float*)d_in[3];  const float* bq = (const float*)d_in[4];
    const float* Wk = (const float*)d_in[5];  const float* bk = (const float*)d_in[6];
    const float* Wv = (const float*)d_in[7];  const float* bv = (const float*)d_in[8];
    const float* Wo = (const float*)d_in[9];  const float* bo = (const float*)d_in[10];

    const size_t NX = (size_t)MROWS * D_EMB;
    const size_t NW = (size_t)D_EMB * D_EMB;
    half_t* xq = (half_t*)d_ws;
    half_t* xk = xq + NX;
    half_t* xv = xk + NX;
    half_t* wq = xv + NX;      // transposed f16 weights [n][k]
    half_t* wk = wq + NW;
    half_t* wv = wk + NW;
    half_t* wo = wv + NW;
    half_t* qp = wo + NW;
    half_t* kp = qp + NX;
    half_t* vt = kp + NX;
    half_t* mm = xq;           // merged ctx reuses xq (dead after Q projection)

    // input conversion + weight transpose/conversion
    cvt4_kernel<<<(int)(NX / 4 / 256), 256, 0, stream>>>((const float4*)query, (v4h*)xq, (int)(NX / 4));
    cvt4_kernel<<<(int)(NX / 4 / 256), 256, 0, stream>>>((const float4*)key,   (v4h*)xk, (int)(NX / 4));
    cvt4_kernel<<<(int)(NX / 4 / 256), 256, 0, stream>>>((const float4*)value, (v4h*)xv, (int)(NX / 4));
    dim3 gt(D_EMB / 32, D_EMB / 32);
    transpose_cvt_kernel<<<gt, 256, 0, stream>>>(Wq, wq);
    transpose_cvt_kernel<<<gt, 256, 0, stream>>>(Wk, wk);
    transpose_cvt_kernel<<<gt, 256, 0, stream>>>(Wv, wv);
    transpose_cvt_kernel<<<gt, 256, 0, stream>>>(Wo, wo);

    // projections (TDM-fed WMMA GEMM)
    dim3 gg(D_EMB / 64, MROWS / 128);   // (16, 32)
    gemm_wmma_kernel<<<gg, 128, 0, stream>>>(xq, wq, bq, qp, MROWS, D_EMB, D_EMB, 0);
    gemm_wmma_kernel<<<gg, 128, 0, stream>>>(xk, wk, bk, kp, MROWS, D_EMB, D_EMB, 0);
    gemm_wmma_kernel<<<gg, 128, 0, stream>>>(xv, wv, bv, vt, MROWS, D_EMB, D_EMB, 1);

    // attention (B*H*(S/16) = 4096 waves, 4 waves/block)
    flash_attn_kernel<<<1024, 128, 0, stream>>>(qp, kp, vt, mm);

    // output projection (f32 + bias)
    gemm_wmma_kernel<<<gg, 128, 0, stream>>>(mm, wo, bo, d_out, MROWS, D_EMB, D_EMB, 2);
}
